// AttnModel_87196426043722
// MI455X (gfx1250) — matrix operand order repaired and checked by reading, not compiled, not run on hardware
//
#include <hip/hip_runtime.h>
#include <hip/hip_bf16.h>
#include <math.h>

// Problem constants (match reference)
constexpr int kB   = 32;
constexpr int kNPG = 4096;
constexpr int kN   = kB * kNPG;   // 131072 nodes
constexpr int kF   = 64;          // F_IN
constexpr int kC   = 128;
constexpr int kK   = 32;
constexpr int kJ   = 40;
constexpr float kEPS = 1e-5f;

typedef __attribute__((ext_vector_type(16))) __bf16 v16bf;
typedef __attribute__((ext_vector_type(8)))  float  v8f;

// ---------------------------------------------------------------------------
// WMMA fragment loaders, A and B in contiguous k order k = 16*(lane>>4) + s (wave32)
// ---------------------------------------------------------------------------

// A-matrix 16x32 bf16 (MxK). lane&15 = row; lane>>4 selects K-half.
// VGPR i: K = 16*kh + 2*i
__device__ __forceinline__ v16bf load_A_tile(const float* __restrict__ base,
                                             int ld, int kvalid) {
  int lane = threadIdx.x & 31;
  int row  = lane & 15;
  int kh   = lane >> 4;
  const float* r = base + (size_t)row * ld;
  v16bf a;
#pragma unroll
  for (int i = 0; i < 8; ++i) {
    int kb = 16 * kh + 2 * i;
    float f0 = (kb     < kvalid) ? r[kb]     : 0.f;
    float f1 = (kb + 1 < kvalid) ? r[kb + 1] : 0.f;
    a[2 * i]     = (__bf16)f0;
    a[2 * i + 1] = (__bf16)f1;
  }
  return a;
}

// B-matrix 32x16 bf16 (KxN), source row-major B(k,n) = base[k*ld + n].
// lane&15 = col; lanes 0-15 hold K=0..15, lanes 16-31 hold K=16..31.
__device__ __forceinline__ v16bf load_B_rm(const float* __restrict__ base,
                                           int ld, int kvalid, int nvalid) {
  int lane = threadIdx.x & 31;
  int col  = lane & 15;
  int kh   = lane >> 4;
  bool cv  = (col < nvalid);
  v16bf b;
#pragma unroll
  for (int i = 0; i < 8; ++i) {
    int k = 16 * kh + 2 * i;
    float f0 = (cv && k     < kvalid) ? base[(size_t)k * ld + col]       : 0.f;
    float f1 = (cv && k + 1 < kvalid) ? base[(size_t)(k + 1) * ld + col] : 0.f;
    b[2 * i]     = (__bf16)f0;
    b[2 * i + 1] = (__bf16)f1;
  }
  return b;
}

// B-matrix 32x16 bf16 (KxN), source transposed: B(k,n) = base[n*ld + k].
__device__ __forceinline__ v16bf load_B_tr(const float* __restrict__ base,
                                           int ld, int kvalid, int nvalid) {
  int lane = threadIdx.x & 31;
  int col  = lane & 15;
  int kh   = lane >> 4;
  bool cv  = (col < nvalid);
  v16bf b;
#pragma unroll
  for (int i = 0; i < 8; ++i) {
    int k = 16 * kh + 2 * i;
    float f0 = (cv && k     < kvalid) ? base[(size_t)col * ld + k]     : 0.f;
    float f1 = (cv && k + 1 < kvalid) ? base[(size_t)col * ld + k + 1] : 0.f;
    b[2 * i]     = (__bf16)f0;
    b[2 * i + 1] = (__bf16)f1;
  }
  return b;
}

__device__ __forceinline__ v8f wmma_bf16(v16bf a, v16bf b, v8f c) {
  return __builtin_amdgcn_wmma_f32_16x16x32_bf16(false, a, false, b,
                                                 (short)0, c, false, false);
}

// C/D 16x16 f32: lane&15 = col; row = r + 8*(lane>>4)
template <bool RELU>
__device__ __forceinline__ void store_C(float* __restrict__ base, int ld,
                                        v8f c, int nvalid) {
  int lane = threadIdx.x & 31;
  int col  = lane & 15;
  int kh   = lane >> 4;
  if (col < nvalid) {
#pragma unroll
    for (int r = 0; r < 8; ++r) {
      float v = c[r];
      if (RELU) v = fmaxf(v, 0.f);
      base[(size_t)(r + 8 * kh) * ld + col] = v;
    }
  }
}

// ---------------------------------------------------------------------------
// K1: out[M x 128] = A[M x 64] @ W[128 x 64]^T + bias ; accumulate sum/sumsq
// Each wave computes 16 rows x 64 cols (4 N-tiles) to keep VGPR pressure low
// (no spills); two waves cover one 16-row strip.
// ---------------------------------------------------------------------------
__global__ void gemm_in_stats(const float* __restrict__ A,
                              const float* __restrict__ W,
                              const float* __restrict__ bias,
                              float* __restrict__ out, int M,
                              float* __restrict__ stats) {
  int gwave = (int)((blockIdx.x * blockDim.x + threadIdx.x) >> 5);
  int mtile = gwave >> 1;            // 16-row strip
  int nhalf = gwave & 1;             // which 64-column half
  int row0  = mtile * 16;
  if (row0 >= M) return;
  int lane = threadIdx.x & 31;
  int col  = lane & 15;
  int kh   = lane >> 4;

  v16bf a0 = load_A_tile(A + (size_t)row0 * kF,      kF, 32);
  v16bf a1 = load_A_tile(A + (size_t)row0 * kF + 32, kF, 32);

  float s1 = 0.f, s2 = 0.f;
#pragma unroll
  for (int t = 0; t < 4; ++t) {
    int n0 = nhalf * 64 + t * 16;
    v16bf b0 = load_B_tr(W + (size_t)n0 * kF,      kF, 32, 16);
    v16bf b1 = load_B_tr(W + (size_t)n0 * kF + 32, kF, 32, 16);
    v8f c = {};
    c = wmma_bf16(a0, b0, c);
    c = wmma_bf16(a1, b1, c);
    float bb = bias[n0 + col];
    float* ob = out + (size_t)row0 * kC + n0;
#pragma unroll
    for (int r = 0; r < 8; ++r) {
      float v = c[r] + bb;
      ob[(size_t)(r + 8 * kh) * kC + col] = v;
      s1 += v;
      s2 += v * v;
    }
  }
#pragma unroll
  for (int off = 16; off > 0; off >>= 1) {
    s1 += __shfl_xor(s1, off, 32);
    s2 += __shfl_xor(s2, off, 32);
  }
  if (lane == 0) {
    atomicAdd(&stats[0], s1);
    atomicAdd(&stats[1], s2);
  }
}

// K2: mean / rsqrt(var) for node and frag layernorms
__global__ void finalize_stats(float* __restrict__ stats) {
  float c1 = (float)kN * (float)kC;
  float m1 = stats[0] / c1;
  float v1 = stats[1] / c1 - m1 * m1;
  stats[4] = m1;
  stats[5] = rsqrtf(v1 + kEPS);
  float c2 = (float)(kB * kJ) * (float)kC;
  float m2 = stats[2] / c2;
  float v2 = stats[3] / c2 - m2 * m2;
  stats[6] = m2;
  stats[7] = rsqrtf(v2 + kEPS);
}

// K2b: x = (x - m) * r, in place, float4
__global__ void norm_inplace(float* __restrict__ X,
                             const float* __restrict__ stats, int n4) {
  int i = (int)(blockIdx.x * blockDim.x + threadIdx.x);
  if (i >= n4) return;
  float m = stats[4], r = stats[5];
  float4 v = ((float4*)X)[i];
  v.x = (v.x - m) * r;
  v.y = (v.y - m) * r;
  v.z = (v.z - m) * r;
  v.w = (v.w - m) * r;
  ((float4*)X)[i] = v;
}

// K3: y_p[row,k] = relu(q[k] * relu(sum_d ysn[row,d] * V[d,k])), 1 wave / row
__global__ void yp_kernel(const float* __restrict__ YSP,
                          const float* __restrict__ V,
                          const float* __restrict__ q,
                          const float* __restrict__ stats,
                          float* __restrict__ YP) {
  int row = blockIdx.x;      // 0..B*J-1
  int k   = threadIdx.x;     // 0..31
  float m = stats[6], r = stats[7];
  const float* ys = YSP + (size_t)row * kC;
  float s = 0.f;
  for (int d = 0; d < kC; ++d)
    s += (ys[d] - m) * r * V[(size_t)d * kK + k];
  s = fmaxf(s, 0.f);
  s = fmaxf(q[k] * s, 0.f);
  YP[(size_t)row * kK + k] = s;
}

// K4: x_p = relu(x @ U), M=kN, K=128, N=32
__global__ void gemm_xp(const float* __restrict__ X,
                        const float* __restrict__ U,
                        float* __restrict__ XP) {
  int gwave = (int)((blockIdx.x * blockDim.x + threadIdx.x) >> 5);
  int row0  = gwave * 16;
  if (row0 >= kN) return;
  v8f c0 = {}, c1 = {};
#pragma unroll
  for (int s = 0; s < 4; ++s) {
    v16bf a  = load_A_tile(X + (size_t)row0 * kC + 32 * s, kC, 32);
    v16bf b0 = load_B_rm(U + (size_t)(32 * s) * kK,      kK, 32, 16);
    v16bf b1 = load_B_rm(U + (size_t)(32 * s) * kK + 16, kK, 32, 16);
    c0 = wmma_bf16(a, b0, c0);
    c1 = wmma_bf16(a, b1, c1);
  }
  store_C<true>(XP + (size_t)row0 * kK,      kK, c0, 16);
  store_C<true>(XP + (size_t)row0 * kK + 16, kK, c1, 16);
}

// K5: I[b] = x_p[b] @ y_p[b]^T, per-graph, N=J=40 (padded to 48)
__global__ void gemm_I(const float* __restrict__ XP,
                       const float* __restrict__ YP,
                       float* __restrict__ Iout) {
  int gwave = (int)((blockIdx.x * blockDim.x + threadIdx.x) >> 5);
  if (gwave >= kB * (kNPG / 16)) return;
  int b  = gwave >> 8;     // kNPG/16 == 256
  int mt = gwave & 255;
  size_t row0 = (size_t)b * kNPG + (size_t)mt * 16;
  v16bf a = load_A_tile(XP + row0 * kK, kK, 32);
  const float* ypb = YP + (size_t)b * kJ * kK;
#pragma unroll
  for (int t = 0; t < 3; ++t) {
    int n0 = t * 16;
    int nv = kJ - n0; if (nv > 16) nv = 16;
    v16bf bt = load_B_tr(ypb + (size_t)n0 * kK, kK, 32, nv);
    v8f c = {};
    c = wmma_bf16(a, bt, c);
    store_C<false>(Iout + row0 * kJ + n0, kJ, c, nv);
  }
}

// K6: softmax over the node axis i (per b, j column)
__global__ void softmax_col(float* __restrict__ Iio) {
  int bj = blockIdx.x;
  int b = bj / kJ, j = bj % kJ;
  float* col = Iio + (size_t)b * kNPG * kJ + j;
  __shared__ float red[256];
  int tid = threadIdx.x;
  float m = -3.4e38f;
  for (int i = tid; i < kNPG; i += 256) m = fmaxf(m, col[(size_t)i * kJ]);
  red[tid] = m; __syncthreads();
  for (int s = 128; s > 0; s >>= 1) {
    if (tid < s) red[tid] = fmaxf(red[tid], red[tid + s]);
    __syncthreads();
  }
  m = red[0]; __syncthreads();
  float sum = 0.f;
  for (int i = tid; i < kNPG; i += 256) sum += __expf(col[(size_t)i * kJ] - m);
  red[tid] = sum; __syncthreads();
  for (int s = 128; s > 0; s >>= 1) {
    if (tid < s) red[tid] += red[tid + s];
    __syncthreads();
  }
  float inv = 1.f / red[0];
  for (int i = tid; i < kNPG; i += 256) {
    size_t o = (size_t)i * kJ;
    col[o] = __expf(col[o] - m) * inv;
  }
}

// K7a / K8: Out[b] = (relu?)(Amat[b] @ y_p[b]); Amat ld=J=40, K padded to 64
template <bool RELU>
__global__ void gemm_Ayp(const float* __restrict__ Amat,
                         const float* __restrict__ YP,
                         float* __restrict__ Out) {
  int gwave = (int)((blockIdx.x * blockDim.x + threadIdx.x) >> 5);
  if (gwave >= kB * (kNPG / 16)) return;
  int b  = gwave >> 8;
  int mt = gwave & 255;
  size_t row0 = (size_t)b * kNPG + (size_t)mt * 16;
  const float* ab  = Amat + row0 * kJ;
  const float* ypb = YP + (size_t)b * kJ * kK;
  v16bf a0 = load_A_tile(ab,      kJ, 32);
  v16bf a1 = load_A_tile(ab + 32, kJ, kJ - 32);   // 8 valid K, rest zero
#pragma unroll
  for (int t = 0; t < 2; ++t) {
    int n0 = t * 16;
    v16bf b0 = load_B_rm(ypb + n0,            kK, 32,       16);
    v16bf b1 = load_B_rm(ypb + 32 * kK + n0,  kK, kJ - 32,  16);
    v8f c = {};
    c = wmma_bf16(a0, b0, c);
    c = wmma_bf16(a1, b1, c);
    store_C<RELU>(Out + row0 * kK + n0, kK, c, 16);
  }
}

// K7b: h_c = h_k @ V^T, M=kN, K=32, N=128 ; V is [C,K] so B(k,c)=V[c*K+k]
__global__ void gemm_hc(const float* __restrict__ HK,
                        const float* __restrict__ Vw,
                        float* __restrict__ HC) {
  int gwave = (int)((blockIdx.x * blockDim.x + threadIdx.x) >> 5);
  int mtile = gwave >> 1;
  int nhalf = gwave & 1;
  int row0  = mtile * 16;
  if (row0 >= kN) return;
  v16bf a = load_A_tile(HK + (size_t)row0 * kK, kK, 32);
#pragma unroll
  for (int t = 0; t < 4; ++t) {
    int n0 = nhalf * 64 + t * 16;
    v16bf bt = load_B_tr(Vw + (size_t)n0 * kK, kK, 32, 16);
    v8f c = {};
    c = wmma_bf16(a, bt, c);
    store_C<false>(HC + (size_t)row0 * kC + n0, kC, c, 16);
  }
}

// K7c: gated residual update, one block (128 threads) per node row
__global__ void gate_kernel(float* __restrict__ X,
                            const float* __restrict__ HC,
                            const float* __restrict__ gW,
                            const float* __restrict__ gb) {
  int node = blockIdx.x;
  int c    = threadIdx.x;   // 0..127
  size_t o = (size_t)node * kC + c;
  float xv = X[o];
  float hv = HC[o];
  __shared__ float red[128];
  red[c] = xv * gW[c] + hv * gW[kC + c];
  __syncthreads();
  for (int s = 64; s > 0; s >>= 1) {
    if (c < s) red[c] += red[c + s];
    __syncthreads();
  }
  float z = 1.f / (1.f + __expf(-(red[0] + gb[0])));
  X[o] = (1.f - z) * xv + z * hv;
}

// K9: out[i] = sigmoid(sum_k x_p[i,k] * p[i,k])
__global__ void final_out(const float* __restrict__ XP,
                          const float* __restrict__ P,
                          float* __restrict__ out, int n) {
  int i = (int)(blockIdx.x * blockDim.x + threadIdx.x);
  if (i >= n) return;
  const float* a = XP + (size_t)i * kK;
  const float* p = P + (size_t)i * kK;
  float s = 0.f;
#pragma unroll
  for (int k = 0; k < kK; ++k) s += a[k] * p[k];
  out[i] = 1.f / (1.f + __expf(-s));
}

// ---------------------------------------------------------------------------
extern "C" void kernel_launch(void* const* d_in, const int* in_sizes, int n_in,
                              void* d_out, int out_size, void* d_ws,
                              size_t ws_size, hipStream_t stream) {
  const float* node_feats = (const float*)d_in[0];
  const float* frag_emb   = (const float*)d_in[1];
  const float* W_in       = (const float*)d_in[2];
  const float* b_in       = (const float*)d_in[3];
  const float* U          = (const float*)d_in[4];
  const float* V          = (const float*)d_in[5];
  const float* q          = (const float*)d_in[6];
  const float* gate_W     = (const float*)d_in[7];
  const float* gate_b     = (const float*)d_in[8];
  float* out = (float*)d_out;
  float* ws  = (float*)d_ws;

  // Workspace layout (floats). Total ~47.4M floats (~190 MB).
  float* stats = ws;                                   // [0..7] + pad
  float* X     = ws + 16;                              // [N, C]
  float* YSP   = X    + (size_t)kN * kC;               // [B*J, C]
  float* YP    = YSP  + (size_t)kB * kJ * kC;          // [B*J, K]
  float* XP    = YP   + (size_t)kB * kJ * kK;          // [N, K]
  float* Ibuf  = XP   + (size_t)kN * kK;               // [N, J]
  float* HK    = Ibuf + (size_t)kN * kJ;               // [N, K] (also p)
  float* HC    = HK   + (size_t)kN * kK;               // [N, C]

  hipMemsetAsync(stats, 0, 4 * sizeof(float), stream);

  // Input projections + global-layernorm statistics.
  // gemm_in_stats: 2 waves per 16-row strip (N split in halves).
  gemm_in_stats<<<(kN / 16) * 2 / 4, 128, 0, stream>>>(node_feats, W_in, b_in,
                                                       X, kN, stats);
  gemm_in_stats<<<((kB * kJ / 16) * 2 + 3) / 4, 128, 0, stream>>>(
      frag_emb, W_in, b_in, YSP, kB * kJ, stats + 2);
  finalize_stats<<<1, 1, 0, stream>>>(stats);
  norm_inplace<<<(kN * kC / 4 + 255) / 256, 256, 0, stream>>>(X, stats,
                                                              kN * kC / 4);
  yp_kernel<<<kB * kJ, 32, 0, stream>>>(YSP, V, q, stats, YP);

  for (int il = 0; il < 3; ++il) {
    gemm_xp<<<kN / 16 / 4, 128, 0, stream>>>(X, U, XP);
    gemm_I<<<kN / 16 / 4, 128, 0, stream>>>(XP, YP, Ibuf);
    if (il < 2) {
      softmax_col<<<kB * kJ, 256, 0, stream>>>(Ibuf);
      gemm_Ayp<true><<<kN / 16 / 4, 128, 0, stream>>>(Ibuf, YP, HK);
      gemm_hc<<<(kN / 16) * 2 / 4, 128, 0, stream>>>(HK, V, HC);
      gate_kernel<<<kN, 128, 0, stream>>>(X, HC, gate_W, gate_b);
    }
  }

  // p = I @ y_p (no relu), then out = sigmoid(<x_p, p>)
  gemm_Ayp<false><<<kN / 16 / 4, 128, 0, stream>>>(Ibuf, YP, HK);
  final_out<<<(kN + 255) / 256, 256, 0, stream>>>(XP, HK, out, kN);
}
